// trainerGauss_25821343384370
// MI455X (gfx1250) — matrix operand order repaired and checked by reading, not compiled, not run on hardware
//
#include <hip/hip_runtime.h>

// ---------------------------------------------------------------------------
// CDNA5 (gfx1250) Wasserstein-barycenter style transfer
//   d = 512, n = 65536 samples (rows are 512 contiguous floats)
// Pipeline:
//   1) mean_kernel       : column means
//   2) cov_kernel        : cov = Xc^T Xc, bf16-split WMMA, split-K atomics
//   3) bary_solver       : persistent 1-block 50-iter barycenter fixed point,
//                          Newton-Schulz matrix sqrt (all WMMA GEMM)
//   4) transform_kernel  : out = (x - mu_c) @ M^T + mu_b via WMMA
// LDS staging is done in WMMA *fragment order* so each lane reads its whole
// 16-element bf16 fragment as 32 contiguous bytes (2x ds_load_b128).
// ---------------------------------------------------------------------------

#define D      512
#define MSZ    (D * D)

typedef __attribute__((ext_vector_type(16))) __bf16 v16bf;
typedef __attribute__((ext_vector_type(8)))  float  v8f;

union BF16x16 {
    v16bf v;
    unsigned short u[16];
    __device__ BF16x16() {}
};

// Split fp32 into bf16 hi + bf16 lo (truncation split; residual ~2^-16 rel).
__device__ __forceinline__ void splitbf(float f, unsigned short& h, unsigned short& l) {
    unsigned u = __float_as_uint(f);
    h = (unsigned short)(u >> 16);
    float fh = __uint_as_float(u & 0xFFFF0000u);
    l = (unsigned short)(__float_as_uint(f - fh) >> 16);
}

__device__ __forceinline__ v8f wmma_bf16(v16bf a, v16bf b, v8f c) {
    // 8 args: (neg_a, A, neg_b, B, c_mod, C, reuse_a, reuse_b)
    return __builtin_amdgcn_wmma_f32_16x16x32_bf16(false, a, false, b,
                                                   (short)0, c, false, false);
}

__device__ __forceinline__ void atomic_add_f32(float* p, float v) {
    __hip_atomic_fetch_add(p, v, __ATOMIC_RELAXED, __HIP_MEMORY_SCOPE_AGENT);
}

// Inverse of A-fragment K mapping (16x32 bf16 A, ISA 7.12.2):
//   lane = m + 16*h ; element e ; K = e + 8*h + (e>=8 ? 8 : 0)
// Given kk in [0,32): g = kk>>3 ; h = g&1 ; e = kk - 8*((g+1)>>1)
__device__ __forceinline__ void a_inv(int kk, int& h, int& e) {
    int g = kk >> 3;
    h = g & 1;
    e = kk - 8 * ((g + 1) >> 1);
}
// Inverse of B-fragment K mapping (32x16 bf16 B): same as A, lane = n + 16*h
__device__ __forceinline__ void b_inv(int kk, int& h, int& e) {
    int g = kk >> 3;
    h = g & 1;
    e = kk - 8 * ((g + 1) >> 1);
}

__device__ __forceinline__ v16bf lds_frag(const unsigned short* p) {
    return *(const v16bf*)p;
}

// ---------------------------------------------------------------------------
// 1) Column means
// ---------------------------------------------------------------------------
__global__ __launch_bounds__(512) void mean_kernel(const float* __restrict__ x,
                                                   float* __restrict__ mean,
                                                   int nsamp, float inv_n) {
    float s = 0.f;
    for (int r = blockIdx.x; r < nsamp; r += gridDim.x)
        s += x[(size_t)r * D + threadIdx.x];
    atomic_add_f32(&mean[threadIdx.x], s * inv_n);
}

// ---------------------------------------------------------------------------
// 2) Covariance GEMM (unscaled): 64x64 tile, 8 waves (4x2 of 16x32), split-K.
// ---------------------------------------------------------------------------
__global__ __launch_bounds__(256) void cov_kernel(const float* __restrict__ x,
                                                  const float* __restrict__ mean,
                                                  float* __restrict__ cov,
                                                  int kPerBlock) {
    // fragment-ordered staging: [rowblock][lane][element]
    __shared__ __align__(32) unsigned short AfH[4][32][16], AfL[4][32][16],
                                            BfH[4][32][16], BfL[4][32][16];
    const int i0 = blockIdx.x * 64, j0 = blockIdx.y * 64;
    const int kStart = blockIdx.z * kPerBlock;
    const int t = threadIdx.x;
    const int w = t >> 5, l = t & 31, lm = l & 15, h = l >> 4;
    const int wr = (w >> 1) * 16;   // wave row-block base (0,16,32,48)
    const int wc = (w & 1) * 32;    // wave col-block base (0,32)

    v8f acc0 = {}; v8f acc1 = {};

    for (int k0 = kStart; k0 < kStart + kPerBlock; k0 += 32) {
        for (int q = t; q < 2048; q += 256) {
            int kk = q >> 6, ii = q & 63;            // coalesced over ii
            size_t row = (size_t)(k0 + kk) * D;
            float av = x[row + i0 + ii] - mean[i0 + ii];
            int ha, ea; a_inv(kk, ha, ea);
            splitbf(av, AfH[ii >> 4][(ii & 15) + 16 * ha][ea],
                        AfL[ii >> 4][(ii & 15) + 16 * ha][ea]);
            float bv = x[row + j0 + ii] - mean[j0 + ii];
            int hb, eb; b_inv(kk, hb, eb);
            splitbf(bv, BfH[ii >> 4][(ii & 15) + 16 * hb][eb],
                        BfL[ii >> 4][(ii & 15) + 16 * hb][eb]);
        }
        if (k0 + 32 < kStart + kPerBlock)
            __builtin_prefetch(&x[(size_t)(k0 + 32 + (t >> 3)) * D + i0 + ((t & 7) << 3)], 0, 0);
        __syncthreads();

        v16bf ah = lds_frag(AfH[wr >> 4][l]);
        v16bf al = lds_frag(AfL[wr >> 4][l]);
        int cb = wc >> 4;
        v16bf bh0 = lds_frag(BfH[cb][l]),     bl0 = lds_frag(BfL[cb][l]);
        v16bf bh1 = lds_frag(BfH[cb + 1][l]), bl1 = lds_frag(BfL[cb + 1][l]);

        acc0 = wmma_bf16(ah, bh0, acc0);
        acc0 = wmma_bf16(ah, bl0, acc0);
        acc0 = wmma_bf16(al, bh0, acc0);
        acc1 = wmma_bf16(ah, bh1, acc1);
        acc1 = wmma_bf16(ah, bl1, acc1);
        acc1 = wmma_bf16(al, bh1, acc1);
        __syncthreads();
    }

    #pragma unroll
    for (int r = 0; r < 8; ++r) {
        int m = i0 + wr + r + 8 * h;
        atomic_add_f32(&cov[(size_t)m * D + j0 + wc + lm],      acc0[r]);
        atomic_add_f32(&cov[(size_t)m * D + j0 + wc + 16 + lm], acc1[r]);
    }
}

// ---------------------------------------------------------------------------
// 3) Persistent single-block solver (1024 threads / 32 waves)
// ---------------------------------------------------------------------------
__device__ float blk_frob2(const float* A) {
    __shared__ float red[32];
    float s = 0.f;
    for (int i = threadIdx.x; i < MSZ; i += 1024) { float v = A[i]; s += v * v; }
    for (int o = 16; o > 0; o >>= 1) s += __shfl_down(s, o, 32);
    if ((threadIdx.x & 31) == 0) red[threadIdx.x >> 5] = s;
    __syncthreads();
    if (threadIdx.x < 32) {
        float v = red[threadIdx.x];
        for (int o = 16; o > 0; o >>= 1) v += __shfl_down(v, o, 32);
        if (threadIdx.x == 0) red[0] = v;
    }
    __syncthreads();
    float r = red[0];
    __syncthreads();
    return r;
}

// C = alpha*A@B (+ beta*Cin), 512x512. 128x128 macro-tiles, 32 waves
// (8 row-blocks x 4 col-blocks, 16x32 per wave), fragment-ordered LDS staging.
__device__ void blk_gemm(const float* A, const float* B, const float* Cin,
                         float* C, float alpha, float beta) {
    __shared__ __align__(32) unsigned short AfH[8][32][16], AfL[8][32][16],
                                            BfH[8][32][16], BfL[8][32][16];
    const int t = threadIdx.x;
    const int w = t >> 5, l = t & 31, lm = l & 15, h = l >> 4;
    const int wrb = w >> 2;   // 0..7 : 16-row block
    const int wcb = w & 3;    // 0..3 : 32-col block
    __syncthreads();
    for (int mt = 0; mt < 16; ++mt) {
        const int m0 = (mt >> 2) * 128, n0 = (mt & 3) * 128;
        v8f acc0 = {}; v8f acc1 = {};
        for (int k0 = 0; k0 < D; k0 += 32) {
            for (int q = t; q < 4096; q += 1024) {
                int kk = q & 31, mm = q >> 5;         // A: coalesced over kk
                float av = A[(size_t)(m0 + mm) * D + k0 + kk];
                int ha, ea; a_inv(kk, ha, ea);
                splitbf(av, AfH[mm >> 4][(mm & 15) + 16 * ha][ea],
                            AfL[mm >> 4][(mm & 15) + 16 * ha][ea]);
            }
            for (int q = t; q < 4096; q += 1024) {
                int jj = q & 127, kk = q >> 7;        // B: coalesced over jj
                float bv = B[(size_t)(k0 + kk) * D + n0 + jj];
                int hb, eb; b_inv(kk, hb, eb);
                splitbf(bv, BfH[jj >> 4][(jj & 15) + 16 * hb][eb],
                            BfL[jj >> 4][(jj & 15) + 16 * hb][eb]);
            }
            __syncthreads();
            v16bf ah = lds_frag(AfH[wrb][l]);
            v16bf al = lds_frag(AfL[wrb][l]);
            int cb = wcb * 2;
            v16bf bh0 = lds_frag(BfH[cb][l]),     bl0 = lds_frag(BfL[cb][l]);
            v16bf bh1 = lds_frag(BfH[cb + 1][l]), bl1 = lds_frag(BfL[cb + 1][l]);
            acc0 = wmma_bf16(ah, bh0, acc0);
            acc0 = wmma_bf16(ah, bl0, acc0);
            acc0 = wmma_bf16(al, bh0, acc0);
            acc1 = wmma_bf16(ah, bh1, acc1);
            acc1 = wmma_bf16(ah, bl1, acc1);
            acc1 = wmma_bf16(al, bh1, acc1);
            __syncthreads();
        }
        #pragma unroll
        for (int r = 0; r < 8; ++r) {
            int m  = m0 + wrb * 16 + r + 8 * h;
            int n_ = n0 + wcb * 32 + lm;
            float v0 = alpha * acc0[r];
            float v1 = alpha * acc1[r];
            if (beta != 0.f) {
                v0 += beta * Cin[(size_t)m * D + n_];
                v1 += beta * Cin[(size_t)m * D + n_ + 16];
            }
            C[(size_t)m * D + n_]      = v0;
            C[(size_t)m * D + n_ + 16] = v1;
        }
    }
    __syncthreads();
}

__device__ void blk_copy_scaled(float* dst, const float* src, float s) {
    for (int i = threadIdx.x; i < MSZ; i += 1024) dst[i] = src[i] * s;
    __syncthreads();
}
__device__ void blk_zero(float* p) {
    for (int i = threadIdx.x; i < MSZ; i += 1024) p[i] = 0.f;
    __syncthreads();
}
__device__ void blk_identity(float* p) {
    for (int i = threadIdx.x; i < MSZ; i += 1024)
        p[i] = ((i >> 9) == (i & 511)) ? 1.f : 0.f;
    __syncthreads();
}
// W = 1.5*I - 0.5*W (in place)
__device__ void blk_newton_T(float* W) {
    for (int i = threadIdx.x; i < MSZ; i += 1024)
        W[i] = (((i >> 9) == (i & 511)) ? 1.5f : 0.f) - 0.5f * W[i];
    __syncthreads();
}
// dst = 0.5*(src + src^T)
__device__ void blk_symmetrize(float* dst, const float* src) {
    for (int i = threadIdx.x; i < MSZ; i += 1024) {
        int r = i >> 9, c = i & 511;
        dst[i] = 0.5f * (src[i] + src[((size_t)c << 9) | r]);
    }
    __syncthreads();
}

// Coupled Newton-Schulz: Ssq_out = sqrt(A), Sis_out = A^{-1/2}.
__device__ void blk_sqrt_invsqrt(const float* A, float* Ssq_out, float* Sis_out,
                                 float* Y, float* Z, float* W, float* Y2, float* Z2) {
    float fro = sqrtf(blk_frob2(A));
    fro = fmaxf(fro, 1e-30f);
    blk_copy_scaled(Y, A, 1.f / fro);
    blk_identity(Z);
    float* y = Y; float* z = Z; float* y2 = Y2; float* z2 = Z2;
    for (int it = 0; it < 16; ++it) {
        blk_gemm(z, y, nullptr, W, 1.f, 0.f);   // W = Z@Y
        blk_newton_T(W);                        // W = 1.5I - 0.5*Z@Y
        blk_gemm(y, W, nullptr, y2, 1.f, 0.f);  // Y' = Y@T
        blk_gemm(W, z, nullptr, z2, 1.f, 0.f);  // Z' = T@Z
        float* tp;
        tp = y; y = y2; y2 = tp;
        tp = z; z = z2; z2 = tp;
    }
    float sq = sqrtf(fro);
    blk_copy_scaled(Ssq_out, y, sq);
    blk_copy_scaled(Sis_out, z, 1.f / sq);
}

__global__ __launch_bounds__(1024) void bary_solver(float* ws, float inv_nm1) {
    float* mean_c = ws;
    float* mean_s = ws + 512;
    float* mean_b = ws + 1024;
    float* mats   = ws + 2048;
    float* cov_c = mats + 0 * MSZ;
    float* cov_s = mats + 1 * MSZ;
    float* SIG   = mats + 2 * MSZ;
    float* ACC   = mats + 3 * MSZ;   // final transform matrix M lands here
    float* SSQ   = mats + 4 * MSZ;
    float* SIS   = mats + 5 * MSZ;
    float* MID   = mats + 6 * MSZ;
    float* MSQ   = mats + 7 * MSZ;
    float* Y     = mats + 8 * MSZ;
    float* Z     = mats + 9 * MSZ;
    float* W     = mats + 10 * MSZ;
    float* Y2    = mats + 11 * MSZ;
    float* Z2    = mats + 12 * MSZ;

    for (int i = threadIdx.x; i < MSZ; i += 1024) {
        cov_c[i] *= inv_nm1;
        cov_s[i] *= inv_nm1;
    }
    if (threadIdx.x < 512)
        mean_b[threadIdx.x] = 0.5f * (mean_c[threadIdx.x] + mean_s[threadIdx.x]);
    __syncthreads();

    blk_copy_scaled(SIG, cov_c, 1.f);   // init with content covariance

    for (int iter = 0; iter < 50; ++iter) {
        blk_sqrt_invsqrt(SIG, SSQ, SIS, Y, Z, W, Y2, Z2);
        blk_zero(ACC);
        for (int j = 0; j < 2; ++j) {
            const float* cj = j ? cov_s : cov_c;
            blk_gemm(SSQ, cj, nullptr, W, 1.f, 0.f);   // W   = S^1/2 @ C_j
            blk_gemm(W, SSQ, nullptr, MID, 1.f, 0.f);  // MID = S^1/2 C_j S^1/2
            blk_sqrt_invsqrt(MID, MSQ, MID, Y, Z, W, Y2, Z2);
            blk_gemm(SIS, MSQ, nullptr, W, 1.f, 0.f);  // W   = S^-1/2 @ MID^1/2
            blk_gemm(W, SIS, ACC, ACC, 0.5f, 1.f);     // ACC += lam * (...)
        }
        blk_symmetrize(SIG, ACC);
    }

    // M = Sc^-1/2 * sqrt(Sc^1/2 SIG Sc^1/2) * Sc^-1/2   (into ACC)
    blk_sqrt_invsqrt(cov_c, SSQ, SIS, Y, Z, W, Y2, Z2);
    blk_gemm(SSQ, SIG, nullptr, W, 1.f, 0.f);
    blk_gemm(W, SSQ, nullptr, MID, 1.f, 0.f);
    blk_sqrt_invsqrt(MID, MSQ, MID, Y, Z, W, Y2, Z2);
    blk_gemm(SIS, MSQ, nullptr, W, 1.f, 0.f);
    blk_gemm(W, SIS, nullptr, ACC, 1.f, 0.f);
}

// ---------------------------------------------------------------------------
// 4) out[r,i] = sum_j (x[r,j]-mu_c[j]) * M[i,j] + mu_b[i]
// ---------------------------------------------------------------------------
__global__ __launch_bounds__(256) void transform_kernel(const float* __restrict__ x,
                                                        const float* __restrict__ mean_c,
                                                        const float* __restrict__ M,
                                                        const float* __restrict__ mean_b,
                                                        float* __restrict__ out) {
    __shared__ __align__(32) unsigned short AfH[4][32][16], AfL[4][32][16],
                                            BfH[4][32][16], BfL[4][32][16];
    const int r0 = blockIdx.x * 64;  // sample rows
    const int i0 = blockIdx.y * 64;  // output cols
    const int t = threadIdx.x;
    const int w = t >> 5, l = t & 31, lm = l & 15, h = l >> 4;
    const int wr = (w >> 1) * 16;
    const int wc = (w & 1) * 32;

    v8f acc0 = {}; v8f acc1 = {};

    for (int k0 = 0; k0 < D; k0 += 32) {
        for (int q = t; q < 2048; q += 256) {
            int kk = q & 31, rr = q >> 5;            // A: coalesced over kk
            float av = x[(size_t)(r0 + rr) * D + k0 + kk] - mean_c[k0 + kk];
            int ha, ea; a_inv(kk, ha, ea);
            splitbf(av, AfH[rr >> 4][(rr & 15) + 16 * ha][ea],
                        AfL[rr >> 4][(rr & 15) + 16 * ha][ea]);
        }
        for (int q = t; q < 2048; q += 256) {
            int kk = q & 31, ii = q >> 5;            // B[k][i]=M[i*D+k], coalesced over kk
            float bv = M[(size_t)(i0 + ii) * D + k0 + kk];
            int hb, eb; b_inv(kk, hb, eb);
            splitbf(bv, BfH[ii >> 4][(ii & 15) + 16 * hb][eb],
                        BfL[ii >> 4][(ii & 15) + 16 * hb][eb]);
        }
        if (k0 + 32 < D)
            __builtin_prefetch(&x[(size_t)(r0 + (t >> 2)) * D + k0 + 32], 0, 0);
        __syncthreads();

        v16bf ah = lds_frag(AfH[wr >> 4][l]);
        v16bf al = lds_frag(AfL[wr >> 4][l]);
        int cb = wc >> 4;
        v16bf bh0 = lds_frag(BfH[cb][l]),     bl0 = lds_frag(BfL[cb][l]);
        v16bf bh1 = lds_frag(BfH[cb + 1][l]), bl1 = lds_frag(BfL[cb + 1][l]);

        acc0 = wmma_bf16(ah, bh0, acc0);
        acc0 = wmma_bf16(ah, bl0, acc0);
        acc0 = wmma_bf16(al, bh0, acc0);
        acc1 = wmma_bf16(ah, bh1, acc1);
        acc1 = wmma_bf16(ah, bl1, acc1);
        acc1 = wmma_bf16(al, bh1, acc1);
        __syncthreads();
    }

    #pragma unroll
    for (int r = 0; r < 8; ++r) {
        int m = r0 + wr + r + 8 * h;
        int n0 = i0 + wc + lm, n1 = i0 + wc + 16 + lm;
        out[(size_t)m * D + n0] = acc0[r] + mean_b[n0];
        out[(size_t)m * D + n1] = acc1[r] + mean_b[n1];
    }
}

// ---------------------------------------------------------------------------
extern "C" void kernel_launch(void* const* d_in, const int* in_sizes, int n_in,
                              void* d_out, int out_size, void* d_ws, size_t ws_size,
                              hipStream_t stream) {
    const float* content = (const float*)d_in[0];
    const float* style   = (const float*)d_in[1];
    // d_in[2] = level (unused by the math)

    const int nsamp = in_sizes[0] / D;      // 65536
    float* ws = (float*)d_ws;
    float* mean_c = ws;
    float* mean_s = ws + 512;
    float* mean_b = ws + 1024;
    float* mats   = ws + 2048;
    float* cov_c  = mats;
    float* cov_s  = mats + MSZ;
    float* Mmat   = mats + 3 * MSZ;         // ACC in solver holds final M
    float* out    = (float*)d_out;

    // zero means + both covariance accumulators
    hipMemsetAsync(ws, 0, (size_t)(2048 + 2 * MSZ) * sizeof(float), stream);

    const float inv_n = 1.f / (float)nsamp;
    mean_kernel<<<1024, 512, 0, stream>>>(content, mean_c, nsamp, inv_n);
    mean_kernel<<<1024, 512, 0, stream>>>(style,   mean_s, nsamp, inv_n);

    const int splitK = 64;
    const int kPerBlock = nsamp / splitK;   // 1024
    cov_kernel<<<dim3(D / 64, D / 64, splitK), 256, 0, stream>>>(content, mean_c, cov_c, kPerBlock);
    cov_kernel<<<dim3(D / 64, D / 64, splitK), 256, 0, stream>>>(style,   mean_s, cov_s, kPerBlock);

    bary_solver<<<1, 1024, 0, stream>>>(ws, 1.f / (float)(nsamp - 1));

    transform_kernel<<<dim3(nsamp / 64, D / 64), 256, 0, stream>>>(content, mean_c, Mmat, mean_b, out);
}
